// Temp_Attn_83236466196699
// MI455X (gfx1250) — matrix operand order repaired and checked by reading, not compiled, not run on hardware
//
#include <hip/hip_runtime.h>
#include <hip/hip_bf16.h>

typedef __attribute__((ext_vector_type(16))) _Float16 v16h;
typedef __attribute__((ext_vector_type(8)))  _Float16 v8h;
typedef __attribute__((ext_vector_type(8)))  float    v8f;
typedef __attribute__((ext_vector_type(4)))  float    v4f;

// Problem constants
#define NB 32
#define NC 64
#define NT 8192
#define LP 1024   // padded L (L=1023)
#define CK 1024   // C*K = 64*16

// ---------------------------------------------------------------------------
// WMMA fragment loaders (gfx1250 wave32; contiguous-k order used for both A and B, not the ISA table order)
// A (16xK f16, row-major source, leading dim `ld` in halves):
//   lane r=lane&15 -> row M=r; 16 contiguous K values per half-wave.
// B (Kx16 f16, source stored transposed as N x K, leading dim `ld`):
//   lane n=lane&15 -> col N=n; 16 contiguous K values per half-wave.
// ---------------------------------------------------------------------------
static __device__ inline v16h load_a_f16(const _Float16* base, int ld, int k0) {
  const int lane = threadIdx.x & 31;
  const int r = lane & 15;
  const int kb = (lane >> 4) << 4;
  const _Float16* p = base + r * ld + k0 + kb;
  v16h out;
  *(v8h*)&out       = *(const v8h*)p;
  *((v8h*)&out + 1) = *(const v8h*)(p + 8);
  return out;
}

static __device__ inline v16h load_b_f16(const _Float16* base, int ld, int k0) {
  const int lane = threadIdx.x & 31;
  const int n = lane & 15;
  const int kb = (lane >> 4) << 4;
  return *(const v16h*)(base + n * ld + k0 + kb);
}

static __device__ inline v8f wmma_f16(v16h a, v16h b, v8f c) {
  return __builtin_amdgcn_wmma_f32_16x16x32_f16(
      /*neg_a=*/false, a, /*neg_b=*/false, b,
      /*c_mod=*/(short)0, c, /*reuse_a=*/false, /*reuse_b=*/false);
}

// ---------------------------------------------------------------------------
// Kernel 0: pack conv weights f32 -> f16, layout W[o][c*16+k] (row-major flat)
// ---------------------------------------------------------------------------
__global__ void pack_weights(const float* __restrict__ wq,
                             const float* __restrict__ wk,
                             const float* __restrict__ wv,
                             _Float16* __restrict__ Wf16) {
  const int i = blockIdx.x * blockDim.x + threadIdx.x;
  const int M = NC * CK;  // 65536
  if (i < M)            Wf16[i] = (_Float16)wq[i];
  else if (i < 2 * M)   Wf16[i] = (_Float16)wk[i - M];
  else if (i < 3 * M)   Wf16[i] = (_Float16)wv[i - 2 * M];
}

// ---------------------------------------------------------------------------
// Kernel 1: strided conv1d as GEMM for q, k, v.
// Grid: (64 l-tiles, 32 batches), 128 threads (4 waves).
// LDS: Pt[16][1024] f16 patch matrix (N x K layout -> B operand).
// Each wave owns 3 of the 12 (matrix, channel-tile) jobs and keeps 3
// accumulators live so each LDS B-fragment feeds 3 WMMAs.
// Outputs: qT,kT as [b][l][c] f16 (padded l->1024), Vh as [b][c][l] f16.
// ---------------------------------------------------------------------------
__global__ __launch_bounds__(128) void conv_qkv(
    const float* __restrict__ x, const _Float16* __restrict__ Wf16,
    const float* __restrict__ bq, const float* __restrict__ bk,
    const float* __restrict__ bv,
    _Float16* __restrict__ qT, _Float16* __restrict__ kT,
    _Float16* __restrict__ Vh) {
  __shared__ _Float16 Pt[16 * CK];  // 32 KB

  const int ltile = blockIdx.x;
  const int b = blockIdx.y;
  const int l0 = ltile * 16;
  const int tid = threadIdx.x;

  // Build patch matrix: Pt[l][c*16+k] = x[b][c][(l0+l)*8 + k]
  for (int it = 0; it < 8; ++it) {
    const int p = tid + it * 128;      // 0..1023 -> (l, c)
    const int l = p >> 6;
    const int c = p & 63;
    _Float16* dst = &Pt[l * CK + c * 16];
    const int lg = l0 + l;
    if (lg <= 1022) {
      const float* src = x + (((size_t)b * NC + c) << 13) + lg * 8;
#pragma unroll
      for (int u = 0; u < 4; ++u) {
        v4f xv = *(const v4f*)(src + u * 4);
        dst[u * 4 + 0] = (_Float16)xv.x;
        dst[u * 4 + 1] = (_Float16)xv.y;
        dst[u * 4 + 2] = (_Float16)xv.z;
        dst[u * 4 + 3] = (_Float16)xv.w;
      }
    } else {
#pragma unroll
      for (int u = 0; u < 16; ++u) dst[u] = (_Float16)0.0f;
    }
  }
  __syncthreads();

  const int wave = tid >> 5;
  const int lane = tid & 31;
  const int n = lane & 15;
  const int hi = lane >> 4;

  // Wave's 3 jobs: jb = wave + 4*rep; mat = jb%3, mtile = jb/3 (all 12 covered)
  int mats[3], mtiles[3];
  const _Float16* Ap[3];
#pragma unroll
  for (int rp = 0; rp < 3; ++rp) {
    const int jb = wave + rp * 4;
    mats[rp] = jb % 3;
    mtiles[rp] = jb / 3;
    Ap[rp] = Wf16 + mats[rp] * (NC * CK) + mtiles[rp] * 16 * CK;
  }

  v8f acc0 = {}, acc1 = {}, acc2 = {};
  for (int kk = 0; kk < 32; ++kk) {
    const int k0 = kk * 32;
    v16h bb = load_b_f16(Pt, CK, k0);   // shared across the 3 jobs
    v16h a0 = load_a_f16(Ap[0], CK, k0);
    acc0 = wmma_f16(a0, bb, acc0);
    v16h a1 = load_a_f16(Ap[1], CK, k0);
    acc1 = wmma_f16(a1, bb, acc1);
    v16h a2 = load_a_f16(Ap[2], CK, k0);
    acc2 = wmma_f16(a2, bb, acc2);
  }

#pragma unroll
  for (int rp = 0; rp < 3; ++rp) {
    v8f acc = (rp == 0) ? acc0 : (rp == 1) ? acc1 : acc2;
    const int mat = mats[rp];
    const float* bias = (mat == 0) ? bq : (mat == 1) ? bk : bv;
    const int cbase = mtiles[rp] * 16 + hi * 8;
#pragma unroll
    for (int e = 0; e < 8; ++e) acc[e] += bias[cbase + e];

    const int l = l0 + n;  // D layout: N (=l) over lanes, M (=c) over elems
    if (mat < 2) {
      // transposed store: [b][l][c], 8 contiguous halves per lane
      _Float16* dst = (mat == 0 ? qT : kT) +
                      ((size_t)b * LP + l) * NC + cbase;
      v8h h;
#pragma unroll
      for (int e = 0; e < 8; ++e) h[e] = (_Float16)acc[e];
      *(v8h*)dst = h;
    } else {
      // row-major store: [b][c][l]
      _Float16* dst = Vh + ((size_t)b * NC + cbase) * LP + l;
#pragma unroll
      for (int e = 0; e < 8; ++e) dst[e * LP] = (_Float16)acc[e];
    }
  }
}

// ---------------------------------------------------------------------------
// Kernel 2: attention for one (batch, 16-query tile).
// Scores strip (16 x 1024) via WMMA -> LDS f16, masked softmax in LDS,
// unnormalized exp() becomes the B operand of the output GEMM; 1/sum folded
// into the final store. Grid: (64 i-tiles, 32 batches), 128 threads.
// ---------------------------------------------------------------------------
__global__ __launch_bounds__(128) void attention(
    const _Float16* __restrict__ qT, const _Float16* __restrict__ kT,
    const _Float16* __restrict__ Vh, float* __restrict__ O) {
  __shared__ _Float16 S[16 * LP];   // 32 KB score strip
  __shared__ float red[16 * 8];
  __shared__ float linv[16];

  const int itile = blockIdx.x;
  const int b = blockIdx.y;
  const int i0 = itile * 16;
  const int tid = threadIdx.x;
  const int wave = tid >> 5;
  const int lane = tid & 31;
  const int n = lane & 15;
  const int hi = lane >> 4;

  // --- scores: S[i][j] = sum_c qT[i][c] * kT[j][c], K = 64 ---
  const _Float16* qBase = qT + ((size_t)b * LP + i0) * NC;
  const _Float16* kBase = kT + (size_t)b * LP * NC;
  const v16h qa0 = load_a_f16(qBase, NC, 0);    // invariant across jt
  const v16h qa1 = load_a_f16(qBase, NC, 32);
  for (int jt = wave * 16; jt < wave * 16 + 16; ++jt) {
    const _Float16* kRow = kBase + (size_t)jt * 16 * NC;
    v8f acc = {};
    acc = wmma_f16(qa0, load_b_f16(kRow, NC, 0), acc);
    acc = wmma_f16(qa1, load_b_f16(kRow, NC, 32), acc);
    const int j = jt * 16 + n;
#pragma unroll
    for (int e = 0; e < 8; ++e) S[(hi * 8 + e) * LP + j] = (_Float16)acc[e];
  }
  __syncthreads();

  // --- softmax across j (8 threads per row, j=1023 is padding -> masked) ---
  const int r = tid >> 3;
  const int sub = tid & 7;
  _Float16* row = &S[r * LP];
  if (sub == 7) row[1023] = (_Float16)(-30000.0f);
  float m = -1e30f;
  for (int jj = sub * 128; jj < sub * 128 + 128; ++jj)
    m = fmaxf(m, (float)row[jj]);
  red[r * 8 + sub] = m;
  __syncthreads();
  float mrow = red[r * 8];
#pragma unroll
  for (int t = 1; t < 8; ++t) mrow = fmaxf(mrow, red[r * 8 + t]);
  __syncthreads();
  float sum = 0.0f;
  for (int jj = sub * 128; jj < sub * 128 + 128; ++jj) {
    float eV = __expf((float)row[jj] - mrow);
    sum += eV;
    row[jj] = (_Float16)eV;
  }
  red[r * 8 + sub] = sum;
  __syncthreads();
  if (sub == 0) {
    float s = 0.0f;
#pragma unroll
    for (int t = 0; t < 8; ++t) s += red[r * 8 + t];
    linv[r] = 1.0f / s;
  }
  __syncthreads();

  // --- output GEMM: O[c][i] = sum_j V[c][j] * P[i][j], K = 1024 ---
  const _Float16* A = Vh + ((size_t)b * NC + wave * 16) * LP;
  v8f o = {};
  for (int kk = 0; kk < 32; ++kk) {
    v16h a = load_a_f16(A, LP, kk * 32);
    v16h bb = load_b_f16(S, LP, kk * 32);
    o = wmma_f16(a, bb, o);
  }
  const float sc = linv[n];
  float* dst = O + ((size_t)b * NC + wave * 16 + hi * 8) * LP + i0 + n;
#pragma unroll
  for (int e = 0; e < 8; ++e) dst[e * LP] = o[e] * sc;
}

// ---------------------------------------------------------------------------
// Kernel 3: nearest-index upsample + residual: out[b,c,t] = O[b,c,t*1023>>13]+x
// ---------------------------------------------------------------------------
__global__ void finalize(const float* __restrict__ x,
                         const float* __restrict__ O,
                         float* __restrict__ out) {
  const size_t g = (size_t)blockIdx.x * blockDim.x + threadIdx.x;
  const size_t flat = g * 4;
  const int rowidx = (int)(flat >> 13);   // b*64 + c
  const int t0 = (int)(flat & (NT - 1));
  v4f xv = *(const v4f*)(x + flat);
  const float* orow = O + (size_t)rowidx * LP;
  v4f rs;
  rs.x = orow[((t0 + 0) * 1023) >> 13] + xv.x;
  rs.y = orow[((t0 + 1) * 1023) >> 13] + xv.y;
  rs.z = orow[((t0 + 2) * 1023) >> 13] + xv.z;
  rs.w = orow[((t0 + 3) * 1023) >> 13] + xv.w;
  *(v4f*)(out + flat) = rs;
}

// ---------------------------------------------------------------------------
extern "C" void kernel_launch(void* const* d_in, const int* in_sizes, int n_in,
                              void* d_out, int out_size, void* d_ws,
                              size_t ws_size, hipStream_t stream) {
  (void)in_sizes; (void)n_in; (void)out_size; (void)ws_size;
  const float* x  = (const float*)d_in[0];
  const float* wq = (const float*)d_in[1];
  const float* bq = (const float*)d_in[2];
  const float* wk = (const float*)d_in[3];
  const float* bk = (const float*)d_in[4];
  const float* wv = (const float*)d_in[5];
  const float* bv = (const float*)d_in[6];
  float* out = (float*)d_out;

  // Workspace layout (~20.4 MB total)
  _Float16* Wf16 = (_Float16*)d_ws;                       // 3*64*1024 halves
  _Float16* qT = Wf16 + (size_t)3 * NC * CK;              // [32][1024][64] f16
  _Float16* kT = qT + (size_t)NB * LP * NC;
  _Float16* Vh = kT + (size_t)NB * LP * NC;               // [32][64][1024] f16
  float* O = (float*)(Vh + (size_t)NB * LP * NC);         // [32][64][1024] f32

  pack_weights<<<768, 256, 0, stream>>>(wq, wk, wv, Wf16);
  conv_qkv<<<dim3(64, NB), 128, 0, stream>>>(x, Wf16, bq, bk, bv, qT, kT, Vh);
  attention<<<dim3(64, NB), 128, 0, stream>>>(qT, kT, Vh, O);
  finalize<<<(NB * NC * NT) / (4 * 256), 256, 0, stream>>>(x, O, out);
}
